// InfiniAttention_6236292514609
// MI455X (gfx1250) — matrix operand order repaired and checked by reading, not compiled, not run on hardware
//
#include <hip/hip_runtime.h>
#include <hip/hip_bf16.h>

// MI455X (gfx1250) InfiniAttention pipeline: bf16 WMMA GEMMs with async
// global->LDS double-buffered staging + flash-style segmented attention with
// linear-memory carry. All matrix math through v_wmma_f32_16x16x32_bf16.

typedef __attribute__((ext_vector_type(16))) __bf16 v16bf;
typedef __attribute__((ext_vector_type(8)))  __bf16 v8bf;
typedef __attribute__((ext_vector_type(8)))  float  v8f;
typedef __attribute__((ext_vector_type(4)))  float  v4f;

#define DEV static __device__ __forceinline__

DEV v16bf cat8(v8bf a, v8bf b) {
  return __builtin_shufflevector(a, b, 0,1,2,3,4,5,6,7,8,9,10,11,12,13,14,15);
}

DEV v8f vzero8() {
  v8f z;
#pragma unroll
  for (int i = 0; i < 8; ++i) z[i] = 0.f;
  return z;
}

DEV v8f wmma_bf16(v16bf a, v16bf b, v8f c) {
  // 8 args: (neg_a, A, neg_b, B, c_mod, C, reuse_a, reuse_b)
  return __builtin_amdgcn_wmma_f32_16x16x32_bf16(false, a, false, b, (short)0, c, false, false);
}

// Async DMA: 16 bytes global -> LDS, tracked by ASYNCcnt (ISA 15.18.3 op 98).
// VDST = per-lane LDS byte address, VADDR = per-lane 64-bit global address.
DEV void async_copy_b128(const __bf16* lds_dst, const __bf16* gsrc) {
  uint32_t loff = (uint32_t)(uintptr_t)lds_dst;   // low 32 bits = LDS offset
  asm volatile("global_load_async_to_lds_b128 %0, %1, off"
               :: "v"(loff), "v"(gsrc)
               : "memory");
}
DEV void wait_async() {
  asm volatile("s_wait_asynccnt 0x0" ::: "memory");
}

// A fragment: 16x32 (MxK) tile from row-major source (stride in elements).
// Per ISA: lanes 0-15 hold K {0..7, 16..23}, lanes 16-31 hold K {8..15, 24..31}.
DEV v16bf ldA(const __bf16* buf, int m0, int k0, int stride, int lm, int lh) {
  const __bf16* row = buf + (size_t)(m0 + lm) * stride + k0;
  v8bf lo = *(const v8bf*)(row + lh * 8);
  v8bf hi = *(const v8bf*)(row + 16 + lh * 8);
  return cat8(lo, hi);
}

// B fragment: 32x16 (KxN) tile from N-major / K-contiguous source.
// Same K order as A: lanes 0-15 hold K {0..7, 16..23}, lanes 16-31 hold K {8..15, 24..31}.
DEV v16bf ldB(const __bf16* buf, int n0, int k0, int stride, int lm, int lh) {
  const __bf16* col = buf + (size_t)(n0 + lm) * stride + k0;
  v8bf lo = *(const v8bf*)(col + lh * 8);
  v8bf hi = *(const v8bf*)(col + 16 + lh * 8);
  return cat8(lo, hi);
}

// ---------------------------------------------------------------------------
// fp32 -> bf16 conversion (straight, for x)
// ---------------------------------------------------------------------------
__global__ void cvt_f32_bf16(const float* __restrict__ in, __bf16* __restrict__ out, int n) {
  for (int i = blockIdx.x * blockDim.x + threadIdx.x; i < n; i += gridDim.x * blockDim.x)
    out[i] = (__bf16)in[i];
}

// fp32 KxN -> bf16 NxK (transposed weights so GEMM B staging is a pure copy)
__global__ void cvt_t_f32_bf16(const float* __restrict__ in, __bf16* __restrict__ out, int n) {
  for (int i = blockIdx.x * blockDim.x + threadIdx.x; i < n; i += gridDim.x * blockDim.x) {
    int k = i >> 11, nn = i & 2047;                // K = N = 2048
    out[(size_t)nn * 2048 + k] = (__bf16)in[i];
  }
}

// ---------------------------------------------------------------------------
// RoPE over consecutive channel pairs of the full EMB (reference semantics),
// fp32 in -> rotated bf16 out. gid indexes (b*T+t)*1024 + i, pair c = 2i.
// ---------------------------------------------------------------------------
__global__ void rope_cvt(const float* __restrict__ in, __bf16* __restrict__ out) {
  int gid = blockIdx.x * 256 + threadIdx.x;
  int i = gid & 1023;
  int t = (gid >> 10) & 4095;
  size_t base = (size_t)(gid >> 10) * 2048 + 2 * i;
  // inv = 10000^(-(2i)/2048); ln(10000) = 9.210340371976184
  float fr = (float)t * __expf(-(float)(2 * i) * (9.210340371976184f / 2048.f));
  float sv, cv;
  __sincosf(fr, &sv, &cv);
  float a = in[base], b = in[base + 1];
  out[base]     = (__bf16)(a * cv - b * sv);
  out[base + 1] = (__bf16)(a * sv + b * cv);
}

// ---------------------------------------------------------------------------
// GEMM: C[M,N] = A_bf16[M,K] x BwT_bf16[N,K]^T + bias.
// Block tile 128x128, 8 waves x (16 rows x 128 cols), K-chunk 32.
// Both LDS tiles staged by global_load_async_to_lds_b128, double-buffered:
// the DMA for chunk c+1 overlaps the 8 WMMAs of chunk c.
// flags bit0: write f32 to Cf; bit1: write bf16 to Cb.
// ---------------------------------------------------------------------------
__global__ __launch_bounds__(256) void gemm_rc(
    const __bf16* __restrict__ A, const __bf16* __restrict__ BwT,
    const float* __restrict__ bias,
    float* __restrict__ Cf, __bf16* __restrict__ Cb,
    int M, int N, int K, int flags)
{
  __shared__ __bf16 Asl[2][128 * 32];   // row-major (m-major), stride 32
  __shared__ __bf16 Btl[2][128 * 32];   // N-major, stride 32

  const int tid = threadIdx.x;
  const int wid = tid >> 5, lane = tid & 31, lh = lane >> 4, lm = lane & 15;
  const int m0 = blockIdx.y * 128, n0 = blockIdx.x * 128;

  v8f acc[8];
#pragma unroll
  for (int j = 0; j < 8; ++j) {
    float bvl = bias[n0 + j * 16 + lm];
#pragma unroll
    for (int r = 0; r < 8; ++r) acc[j][r] = bvl;
  }

  auto stage = [&](int kc, int buf) {
#pragma unroll
    for (int p = 0; p < 2; ++p) {
      int idx = tid + p * 256;                 // 0..511: 128 rows x 4 segs of 8
      int r = idx >> 2, sg = (idx & 3) * 8;
      async_copy_b128(&Asl[buf][r * 32 + sg], &A  [(size_t)(m0 + r) * K + kc + sg]);
      async_copy_b128(&Btl[buf][r * 32 + sg], &BwT[(size_t)(n0 + r) * K + kc + sg]);
    }
  };

  const int nch = K / 32;
  stage(0, 0);
  wait_async();
  __syncthreads();

  for (int c = 0; c < nch; ++c) {
    int cur = c & 1;
    if (c + 1 < nch) stage((c + 1) * 32, 1 - cur);   // DMA next chunk
    v16bf af = ldA(Asl[cur], wid * 16, 0, 32, lm, lh);
#pragma unroll
    for (int j = 0; j < 8; ++j) {
      v16bf bf_ = ldB(Btl[cur], j * 16, 0, 32, lm, lh);
      acc[j] = wmma_bf16(af, bf_, acc[j]);
    }
    wait_async();
    __syncthreads();
  }

#pragma unroll
  for (int j = 0; j < 8; ++j) {
#pragma unroll
    for (int r = 0; r < 8; ++r) {
      int m = m0 + wid * 16 + r + lh * 8;
      int n = n0 + j * 16 + lm;
      float v = acc[j][r];
      if (flags & 1) Cf[(size_t)m * N + n] = v;
      if (flags & 2) Cb[(size_t)m * N + n] = (__bf16)v;
    }
  }
}

// ---------------------------------------------------------------------------
// Segmented Infini-attention. One block per (b,h); sequential scan over 8
// segments carrying mem (128x128) and zsum (128) in LDS.
// "Buggy" reshape index map: for row l in segment s of head h:
//   t = h*256 + s*32 + l/16 ; c = (l%16)*128 + d  (d contiguous).
// Emits attn directly in bf16 (feeds the final WMMA GEMM).
// ---------------------------------------------------------------------------
__global__ __launch_bounds__(256) void infini_attn(
    const __bf16* __restrict__ qb, const __bf16* __restrict__ kb,
    const __bf16* __restrict__ vb, const float* __restrict__ beta,
    __bf16* __restrict__ attnb)
{
  __shared__ float  memf[128 * 128];   // mem carry fp32, layout [e*128 + d]
  __shared__ __bf16 memb[128 * 128];   // bf16 copy for WMMA B operand
  __shared__ float  zsum[128];         // z carry (all rows identical in ref)
  __shared__ __bf16 kch [128 * 128];   // k chunk, row-major [l2*128 + d]
  __shared__ __bf16 vT  [128 * 128];   // v chunk transposed [e*128 + l2]
  __shared__ __bf16 skT [128 * 128];   // (elu(k)+1)^T chunk [d*128 + l2]
  __shared__ __bf16 pscr[8 * 16 * 32]; // per-wave P-tile relayout scratch

  const int tid = threadIdx.x;
  const int wid = tid >> 5, lane = tid & 31, lh = lane >> 4, lm = lane & 15;
  const int b = blockIdx.x >> 4, h = blockIdx.x & 15;
  const float g = 1.f / (1.f + __expf(-beta[0]));
  const float scale = 0.08838834764831845f;   // 1/sqrt(128)
  const size_t seqbase = (size_t)(b * 4096 + h * 256) * 2048;

  for (int i = tid; i < 128 * 128; i += 256) { memf[i] = 0.f; memb[i] = (__bf16)0.f; }
  if (tid < 128) zsum[tid] = 0.f;
  __syncthreads();

  for (int s = 0; s < 8; ++s) {
    // ================= Phase 1: attn for all 512 q rows (old mem/z) =========
    for (int qp = 0; qp < 4; ++qp) {
      int lq = qp * 128 + wid * 16 + lm;  // this lane's q row within segment
      size_t rowq = seqbase + (size_t)(s * 32 + (lq >> 4)) * 2048 + (lq & 15) * 128;

      v16bf qa[4], sqa[4];
      float psum = 0.f;
#pragma unroll
      for (int kk = 0; kk < 4; ++kk) {
        v8bf lo = *(const v8bf*)&qb[rowq + kk * 32 + lh * 8];
        v8bf hi = *(const v8bf*)&qb[rowq + kk * 32 + 16 + lh * 8];
        qa[kk] = cat8(lo, hi);
        v8bf sl, sh;
#pragma unroll
        for (int e = 0; e < 8; ++e) {
          float x = (float)lo[e]; float y = x > 0.f ? x + 1.f : __expf(x);
          psum += y; sl[e] = (__bf16)y;
          x = (float)hi[e]; y = x > 0.f ? x + 1.f : __expf(x);
          psum += y; sh[e] = (__bf16)y;
        }
        sqa[kk] = cat8(sl, sh);
      }
      float rowsum = psum + __shfl_xor(psum, 16);  // full elu(q)+1 row sum

      float rmax[8], rsum[8];
      v8f oacc[8];
#pragma unroll
      for (int r = 0; r < 8; ++r) { rmax[r] = -3.0e38f; rsum[r] = 0.f; }
#pragma unroll
      for (int j = 0; j < 8; ++j) oacc[j] = vzero8();

      for (int jc = 0; jc < 4; ++jc) {
        __syncthreads();
        for (int it = 0; it < 8; ++it) {           // stage 128x128 k and v^T
          int idx = tid + it * 256;
          int r = idx >> 4, sg = (idx & 15) * 8;
          int l2 = jc * 128 + r;
          size_t rowk = seqbase + (size_t)(s * 32 + (l2 >> 4)) * 2048 + (l2 & 15) * 128;
          async_copy_b128(&kch[r * 128 + sg], &kb[rowk + sg]);   // DMA, no VGPR
          v8bf vv = *(const v8bf*)&vb[rowk + sg];
#pragma unroll
          for (int e = 0; e < 8; ++e) vT[(sg + e) * 128 + r] = vv[e];
        }
        wait_async();
        __syncthreads();
        for (int ct = 0; ct < 4; ++ct) {           // 32 kv columns per step
          v8f s0 = vzero8(), s1 = vzero8();
#pragma unroll
          for (int kk = 0; kk < 4; ++kk) {         // scores: K = D = 128
            v16bf b0 = ldB(kch, ct * 32,      kk * 32, 128, lm, lh);
            v16bf b1 = ldB(kch, ct * 32 + 16, kk * 32, 128, lm, lh);
            s0 = wmma_bf16(qa[kk], b0, s0);
            s1 = wmma_bf16(qa[kk], b1, s1);
          }
#pragma unroll
          for (int r = 0; r < 8; ++r) {            // online softmax per row
            float x0 = s0[r] * scale, x1 = s1[r] * scale;
            float t0 = x0, t1 = x1;
#pragma unroll
            for (int m = 1; m < 16; m <<= 1) {
              t0 = fmaxf(t0, __shfl_xor(t0, m));
              t1 = fmaxf(t1, __shfl_xor(t1, m));
            }
            float nmax = fmaxf(rmax[r], fmaxf(t0, t1));
            float corr = __expf(rmax[r] - nmax);
            rmax[r] = nmax;
            float p0 = __expf(x0 - nmax), p1 = __expf(x1 - nmax);
            float ps = p0 + p1;
#pragma unroll
            for (int m = 1; m < 16; m <<= 1) ps += __shfl_xor(ps, m);
            rsum[r] = rsum[r] * corr + ps;
#pragma unroll
            for (int j = 0; j < 8; ++j) oacc[j][r] *= corr;
            // D-layout -> A-layout relayout of P via per-wave LDS scratch
            pscr[wid * 512 + (r + lh * 8) * 32 + lm]      = (__bf16)p0;
            pscr[wid * 512 + (r + lh * 8) * 32 + 16 + lm] = (__bf16)p1;
          }
          asm volatile("s_wait_dscnt 0" ::: "memory");
          v16bf pa = ldA(&pscr[wid * 512], 0, 0, 32, lm, lh);
#pragma unroll
          for (int j = 0; j < 8; ++j) {            // O += P (16x32) @ V (32x128)
            v16bf bv = ldB(vT, j * 16, ct * 32, 128, lm, lh);
            oacc[j] = wmma_bf16(pa, bv, oacc[j]);
          }
        }
      }
      // A_mem = sq @ mem  (den = rowsum(sq)*zsum[e] + 1e-6)
      v8f macc[8];
#pragma unroll
      for (int j = 0; j < 8; ++j) macc[j] = vzero8();
#pragma unroll
      for (int kk = 0; kk < 4; ++kk) {
#pragma unroll
        for (int j = 0; j < 8; ++j) {
          v16bf bm = ldB(memb, j * 16, kk * 32, 128, lm, lh);
          macc[j] = wmma_bf16(sqa[kk], bm, macc[j]);
        }
      }
      // combine g*A_mem + (1-g)*A_dot and store (buggy-reshape inverse map)
#pragma unroll
      for (int j = 0; j < 8; ++j) {
        float zs = zsum[j * 16 + lm];
#pragma unroll
        for (int r = 0; r < 8; ++r) {
          float rs = __shfl(rowsum, r + lh * 8);
          float den = rs * zs + 1e-6f;
          float am = macc[j][r] / den;
          float ad = oacc[j][r] / rsum[r];
          float val = g * am + (1.f - g) * ad;
          int m = r + lh * 8;
          int lrow = qp * 128 + wid * 16 + m;
          size_t orow = seqbase + (size_t)(s * 32 + (lrow >> 4)) * 2048
                      + (lrow & 15) * 128 + j * 16 + lm;
          attnb[orow] = (__bf16)val;
        }
      }
    }
    __syncthreads();

    // ================= Phase 2: mem += sk^T @ v ; zsum += colsum(sk) ========
    v8f mc[8];
#pragma unroll
    for (int j = 0; j < 8; ++j) {
#pragma unroll
      for (int r = 0; r < 8; ++r)
        mc[j][r] = memf[(j * 16 + lm) * 128 + wid * 16 + r + lh * 8];
    }
    for (int jc = 0; jc < 4; ++jc) {
      __syncthreads();
      for (int it = 0; it < 8; ++it) {
        int idx = tid + it * 256;
        int r = idx >> 4, sg = (idx & 15) * 8;
        int l2 = jc * 128 + r;
        size_t rowk = seqbase + (size_t)(s * 32 + (l2 >> 4)) * 2048 + (l2 & 15) * 128;
        v8bf kv = *(const v8bf*)&kb[rowk + sg];
        v8bf vv = *(const v8bf*)&vb[rowk + sg];
#pragma unroll
        for (int e = 0; e < 8; ++e) {
          float x = (float)kv[e];
          float y = x > 0.f ? x + 1.f : __expf(x);
          skT[(sg + e) * 128 + r] = (__bf16)y;
          atomicAdd(&zsum[sg + e], y);
          vT[(sg + e) * 128 + r] = vv[e];
        }
      }
      __syncthreads();
#pragma unroll
      for (int kk = 0; kk < 4; ++kk) {             // K = l2 chunk of 128
        v16bf af = ldA(skT, wid * 16, kk * 32, 128, lm, lh);
#pragma unroll
        for (int j = 0; j < 8; ++j) {
          v16bf bv = ldB(vT, j * 16, kk * 32, 128, lm, lh);
          mc[j] = wmma_bf16(af, bv, mc[j]);
        }
      }
    }
    __syncthreads();
#pragma unroll
    for (int j = 0; j < 8; ++j) {
#pragma unroll
      for (int r = 0; r < 8; ++r) {
        int d = wid * 16 + r + lh * 8;
        int e = j * 16 + lm;
        float v = mc[j][r];
        memf[e * 128 + d] = v;
        memb[e * 128 + d] = (__bf16)v;
      }
    }
    __syncthreads();
  }
}

// ---------------------------------------------------------------------------
extern "C" void kernel_launch(void* const* d_in, const int* in_sizes, int n_in,
                              void* d_out, int out_size, void* d_ws, size_t ws_size,
                              hipStream_t stream)
{
  (void)in_sizes; (void)n_in; (void)out_size; (void)ws_size;
  const float* x    = (const float*)d_in[0];
  const float* Wq   = (const float*)d_in[1];
  const float* bq   = (const float*)d_in[2];
  const float* Wk   = (const float*)d_in[3];
  const float* bk   = (const float*)d_in[4];
  const float* Wv   = (const float*)d_in[5];
  const float* bv   = (const float*)d_in[6];
  const float* Wo   = (const float*)d_in[7];
  const float* bo   = (const float*)d_in[8];
  const float* beta = (const float*)d_in[9];

  const int M = 4 * 4096, N = 2048, K = 2048;

  char* w = (char*)d_ws;
  auto take = [&](size_t bytes) -> char* {
    char* p = w;
    w += (bytes + 255) & ~(size_t)255;
    return p;
  };
  __bf16* xb   = (__bf16*)take((size_t)M * K * 2);
  __bf16* wqT  = (__bf16*)take((size_t)2048 * 2048 * 2);
  __bf16* wkT  = (__bf16*)take((size_t)2048 * 2048 * 2);
  __bf16* wvT  = (__bf16*)take((size_t)2048 * 2048 * 2);
  __bf16* woT  = (__bf16*)take((size_t)2048 * 2048 * 2);
  float*  tmpf = (float*) take((size_t)M * N * 4);   // pre-rope q/k (reused)
  __bf16* qbb  = (__bf16*)take((size_t)M * N * 2);
  __bf16* kbb  = (__bf16*)take((size_t)M * N * 2);
  __bf16* vbb  = (__bf16*)take((size_t)M * N * 2);
  __bf16* attnb= (__bf16*)take((size_t)M * N * 2);

  dim3 blk(256);
  cvt_f32_bf16 <<<8192, blk, 0, stream>>>(x,  xb,  M * K);
  cvt_t_f32_bf16<<<4096, blk, 0, stream>>>(Wq, wqT, 2048 * 2048);
  cvt_t_f32_bf16<<<4096, blk, 0, stream>>>(Wk, wkT, 2048 * 2048);
  cvt_t_f32_bf16<<<4096, blk, 0, stream>>>(Wv, wvT, 2048 * 2048);
  cvt_t_f32_bf16<<<4096, blk, 0, stream>>>(Wo, woT, 2048 * 2048);

  dim3 gg(N / 128, M / 128);
  gemm_rc<<<gg, blk, 0, stream>>>(xb, wqT, bq, tmpf, nullptr, M, N, K, 1);
  rope_cvt<<<(M * 1024) / 256, blk, 0, stream>>>(tmpf, qbb);
  gemm_rc<<<gg, blk, 0, stream>>>(xb, wkT, bk, tmpf, nullptr, M, N, K, 1);
  rope_cvt<<<(M * 1024) / 256, blk, 0, stream>>>(tmpf, kbb);
  gemm_rc<<<gg, blk, 0, stream>>>(xb, wvT, bv, nullptr, vbb, M, N, K, 2);

  infini_attn<<<64, blk, 0, stream>>>(qbb, kbb, vbb, beta, attnb);

  gemm_rc<<<gg, blk, 0, stream>>>(attnb, woT, bo, (float*)d_out, nullptr, M, N, K, 1);
}
